// TraditionalDampedOscillator_50148038148156
// MI455X (gfx1250) — compile-verified, matrix operand order checked by reading, not run on hardware
//
#include <hip/hip_runtime.h>
#include <cstdint>

// ---- problem constants (match reference) ----
constexpr int   MODE_NUM = 1024;
constexpr int   S        = 88200;      // samples
constexpr int   F        = 4410;       // force length
constexpr int   ACH      = 8;          // audio channels
constexpr float SR       = 44100.0f;
constexpr float ALPHA    = 3.0f;
constexpr float BETA     = 1e-7f;
constexpr float TWO_PI_F = 6.28318530717958647692f;

// ---- GEMM / layout constants ----
constexpr int KPAD = 4416;             // F padded to multiple of 32 (138*32)
constexpr int KB   = KPAD / 32;        // 138 k-blocks per tile
constexpr int PADL = 4416;             // left zero pad of signal (>= F-1)
constexpr int NBUF = 92672;            // PADL + S + slack (covers max idx 92629)
constexpr int NT   = (S + 15) / 16;    // 5513 output tiles of 16 samples

// ---- workspace layout (bytes) ----
constexpr size_t OFF_PARAMS = 0;                         // 1024 * float4  = 16384
constexpr size_t OFF_FFLIP  = 16384;                     // 16*4416 f16    = 141312
constexpr size_t OFF_BUF1   = OFF_FFLIP + 141312;        // NBUF f16       = 185344
constexpr size_t OFF_BUF2   = OFF_BUF1 + (size_t)NBUF*2; // NBUF f16
constexpr size_t WS_NEED    = OFF_BUF2 + (size_t)NBUF*2; // 528384 B

typedef __attribute__((ext_vector_type(16))) _Float16     v16h;
typedef __attribute__((ext_vector_type(8)))  _Float16     v8h;
typedef __attribute__((ext_vector_type(8)))  float        v8f;
typedef __attribute__((ext_vector_type(8)))  unsigned int v8u;

// -------- zero-init workspace (padding must be zero each call) --------
__global__ void k_zero(uint32_t* p, int n) {
    for (int i = blockIdx.x * blockDim.x + threadIdx.x; i < n; i += gridDim.x * blockDim.x)
        p[i] = 0u;
}

// -------- per-mode constants: d0, w, and rotation step g*exp(i*dTheta) --------
__global__ void k_modes(const float* __restrict__ freq, float4* __restrict__ mp) {
    int m = blockIdx.x * blockDim.x + threadIdx.x;
    if (m >= MODE_NUM) return;
    float w0  = freq[m] * TWO_PI_F;
    float lbd = w0 * w0;
    float d0  = 0.5f * (ALPHA + BETA * lbd);
    float w   = sqrtf(lbd - d0 * d0);          // = 2*pi*freq0
    float g   = __expf(-d0 * (1.0f / SR));     // per-sample decay
    float sn, cs;
    __sincosf(w * (1.0f / SR), &sn, &cs);      // per-sample phase step
    mp[m] = make_float4(d0, w, g * cs, g * sn);
}

// -------- flipped forces -> f16, padded [16][KPAD] (rows 8..15 and j>=F zero) --------
__global__ void k_fflip(const float* __restrict__ forces, _Float16* __restrict__ ff) {
    int i = blockIdx.x * blockDim.x + threadIdx.x;
    if (i >= 16 * KPAD) return;
    int m = i / KPAD, j = i - m * KPAD;
    float v = 0.0f;
    if (m < ACH && j < F) v = forces[m * F + (F - 1 - j)];
    ff[i] = (_Float16)v;
}

// -------- modal synthesis: recurrence over R=4 samples per thread --------
__global__ void k_synth(const float4* __restrict__ mp,
                        _Float16* __restrict__ buf1, _Float16* __restrict__ buf2) {
    int tid = blockIdx.x * blockDim.x + threadIdx.x;
    if (tid >= S / 4) return;
    int s0 = tid * 4;
    float t1 = (float)(s0 + 1) * (1.0f / SR);
    float a0 = 0.f, a1 = 0.f, a2 = 0.f, a3 = 0.f;
    for (int m = 0; m < MODE_NUM; ++m) {          // uniform address -> scalar loads
        float4 c = mp[m];
        float amp = __expf(-c.x * t1);
        float sn, cs;
        __sincosf(c.y * t1, &sn, &cs);
        float p = amp * cs, q = amp * sn;
        a0 += q;
        float np = p * c.z - q * c.w; q = p * c.w + q * c.z; p = np; a1 += q;
        np       = p * c.z - q * c.w; q = p * c.w + q * c.z; p = np; a2 += q;
        np       = p * c.z - q * c.w; q = p * c.w + q * c.z;          a3 += q;
    }
    _Float16 h0 = (_Float16)a0, h1 = (_Float16)a1, h2 = (_Float16)a2, h3 = (_Float16)a3;
    int b = PADL + s0;
    buf1[b] = h0; buf1[b + 1] = h1; buf1[b + 2] = h2; buf1[b + 3] = h3;   // buf1[i] = sig[i-PADL]
    buf2[b - 1] = h0; buf2[b] = h1; buf2[b + 1] = h2; buf2[b + 2] = h3;   // buf2[i] = sig[i+1-PADL]
}

// -------- convolution as Toeplitz GEMM via v_wmma_f32_16x16x32_f16 --------
__global__ void __launch_bounds__(256) k_conv(const _Float16* __restrict__ ff,
                                              const _Float16* __restrict__ buf1,
                                              const _Float16* __restrict__ buf2,
                                              float* __restrict__ out) {
    const int lane = threadIdx.x & 31;
    const int wave = threadIdx.x >> 5;
    const int tile = blockIdx.x * 8 + wave;
    if (tile >= NT) return;                    // wave-uniform exit: EXEC stays full
    const int tbase = tile * 16;
    const int n  = lane & 15;                  // N column / A row (M) for this lane
    const int kh = lane >> 4;                  // K half select

    // A operand: row n, K runs start at 8*kh (16B aligned: row stride 4416 f16)
    const v8h* ap = (const v8h*)(ff + n * KPAD + 8 * kh);

    // B operand: sig index base; pick parity-matched buffer for aligned dword gathers
    const int base = PADL + tbase - (F - 1);
    const int ib0  = base + n + 8 * kh;        // element start for k0 = 0 (run 1)
    const int p    = ib0 & 1;
    const unsigned int* bp0 =
        (const unsigned int*)(p ? buf2 : buf1) + ((ib0 - p) >> 1);

    v8f c = {};
    for (int kb = 0; kb < KB; ++kb) {
        const int k0 = kb * 32;
        v8h alo = ap[(k0 >> 3)];               // A[m][k0+8kh .. +7]
        v8h ahi = ap[(k0 >> 3) + 2];           // A[m][k0+16+8kh .. +7]
        const unsigned int* q = bp0 + (k0 >> 1);
        v8u bu;
        bu[0] = q[0]; bu[1] = q[1]; bu[2] = q[2]; bu[3] = q[3];   // run 1: 8 f16
        bu[4] = q[8]; bu[5] = q[9]; bu[6] = q[10]; bu[7] = q[11]; // run 2: 8 f16 (+16 elems)
        v16h a;
        #pragma unroll
        for (int i = 0; i < 8; ++i) { a[i] = alo[i]; a[8 + i] = ahi[i]; }
        v16h b = __builtin_bit_cast(v16h, bu);
        c = __builtin_amdgcn_wmma_f32_16x16x32_f16(
                /*neg_a=*/false, a, /*neg_b=*/false, b,
                /*c_mod=*/(short)0, c, /*reuse_a=*/false, /*reuse_b=*/false);
    }

    // C layout: lanes 0-15 hold M=0..7 (our 8 channels) in VGPRs 0..7; lanes 16-31 are padding rows.
    if (lane < 16) {
        int t = tbase + n;
        if (t < S) {
            #pragma unroll
            for (int v = 0; v < 8; ++v) out[v * S + t] = c[v];
        }
    }
}

extern "C" void kernel_launch(void* const* d_in, const int* in_sizes, int n_in,
                              void* d_out, int out_size, void* d_ws, size_t ws_size,
                              hipStream_t stream) {
    (void)in_sizes; (void)n_in; (void)out_size; (void)ws_size;
    const float* freq   = (const float*)d_in[0];   // [1024]
    const float* forces = (const float*)d_in[1];   // [8, 4410]
    float* out = (float*)d_out;                    // [8, 88200]

    char* ws = (char*)d_ws;
    float4*   mp = (float4*)  (ws + OFF_PARAMS);
    _Float16* ff = (_Float16*)(ws + OFF_FFLIP);
    _Float16* b1 = (_Float16*)(ws + OFF_BUF1);
    _Float16* b2 = (_Float16*)(ws + OFF_BUF2);

    k_zero <<<256, 256, 0, stream>>>((uint32_t*)ws, (int)(WS_NEED / 4));
    k_modes<<<(MODE_NUM + 255) / 256, 256, 0, stream>>>(freq, mp);
    k_fflip<<<(16 * KPAD + 255) / 256, 256, 0, stream>>>(forces, ff);
    k_synth<<<(S / 4 + 255) / 256, 256, 0, stream>>>(mp, b1, b2);
    k_conv <<<(NT + 7) / 8, 256, 0, stream>>>(ff, b1, b2, out);
}
